// MlaNer1_84344567759385
// MI455X (gfx1250) — matrix operand order repaired and checked by reading, not compiled, not run on hardware
//
#include <hip/hip_runtime.h>
#include <hip/hip_bf16.h>
#include <math.h>

typedef __attribute__((ext_vector_type(16))) _Float16 v16h;
typedef __attribute__((ext_vector_type(8)))  float    v8f;
typedef __attribute__((ext_vector_type(4)))  float    f32x4;   // clang ext-vector (nontemporal-load friendly)

namespace {
constexpr int kBDim = 64, kLDim = 512, kH = 1536, kT = 64;
constexpr int kRows         = kBDim * kLDim;        // 32768
constexpr int kKSteps       = kH / 32;              // 48
constexpr int kWavesPerBlk  = 8;
constexpr int kRowsPerBlk   = 16 * kWavesPerBlk;    // 128
constexpr int kBlocks       = kRows / kRowsPerBlk;  // 256
constexpr int kLdsHalves    = kH * kT;              // 98304 halves = 192 KB
}

// Zero the global accumulators (loss, right, count) each launch.
__global__ void mla_ner_init(float* g) {
    if (threadIdx.x < 3) g[threadIdx.x] = 0.0f;
}

__global__ void __launch_bounds__(256)
mla_ner_fused(const float* __restrict__ x, const float* __restrict__ W,
              const float* __restrict__ bias, const int* __restrict__ tags,
              const float* __restrict__ amask, const float* __restrict__ t2s,
              const int* __restrict__ texp_p, float* __restrict__ gacc) {
    extern __shared__ _Float16 sW[];  // 192 KB: W as f16 in WMMA-B fragment layout

    const int tid = threadIdx.x;

    // ---- Stage W [1536,64] f32 -> LDS f16, B-fragment layout -------------
    // Fragment (kk, c): K rows kk*32..+31, N cols c*16..+15.
// Lane l<16  holds col (c*16+l),      K = kk*32 + {0..7,16..23}
// Lane l>=16 holds col (c*16+l-16),   K = kk*32 + {8..15,24..31}
    // f32x4 loads: n is a multiple of 4 -> all 4 elements share k (kk,kin,h)
    // and the same column-tile c; lanes differ by +0..+3.
    for (int i4 = tid; i4 < (kH * kT) / 4; i4 += 256) {
        const int i = i4 * 4;
        const int k = i >> 6;        // row in H2
        const int n = i & 63;        // col in T (multiple of 4)
        const int kk = k >> 5, kin = k & 31;
        const int c = n >> 4;
        const int lane0 = (n & 15) + ((kin & 8) ? 16 : 0);
        const int h = (kin & 7) + ((kin & 16) ? 8 : 0);
        const f32x4 w = *(const f32x4*)(W + i);
        _Float16* dst = &sW[(((kk << 2) + c) * 32 + lane0) * 16 + h];
        dst[0 * 16] = (_Float16)w[0];
        dst[1 * 16] = (_Float16)w[1];
        dst[2 * 16] = (_Float16)w[2];
        dst[3 * 16] = (_Float16)w[3];
    }
    __syncthreads();

    const int lane = tid & 31;
    const int wv   = tid >> 5;
    const int row_base = blockIdx.x * kRowsPerBlk + wv * 16;
    const int mrow = lane & 15;        // A-row / D-col within half
    const int hi   = lane >> 4;        // which K/M half this lane serves

    // A 16x32 f16 layout: lane<16 holds K{0..7,16..23}, lane>=16 K{8..15,24..31}
    const float* xrow = x + (size_t)(row_base + mrow) * kH + hi * 8;

    v8f acc0 = {}, acc1 = {}, acc2 = {}, acc3 = {};

    for (int kk = 0; kk < kKSteps; ++kk) {
        // Non-temporal: x is a 192MB stream read exactly once; keep W + the
        // small per-row arrays resident in the 192MB L2 instead.
        const float* ap = xrow + kk * 32;
        const f32x4 a0 = __builtin_nontemporal_load((const f32x4*)(ap + 0));
        const f32x4 a1 = __builtin_nontemporal_load((const f32x4*)(ap + 4));
        const f32x4 a2 = __builtin_nontemporal_load((const f32x4*)(ap + 16));
        const f32x4 a3 = __builtin_nontemporal_load((const f32x4*)(ap + 20));
        v16h A;
        A[0]  = (_Float16)a0[0]; A[1]  = (_Float16)a0[1]; A[2]  = (_Float16)a0[2]; A[3]  = (_Float16)a0[3];
        A[4]  = (_Float16)a1[0]; A[5]  = (_Float16)a1[1]; A[6]  = (_Float16)a1[2]; A[7]  = (_Float16)a1[3];
        A[8]  = (_Float16)a2[0]; A[9]  = (_Float16)a2[1]; A[10] = (_Float16)a2[2]; A[11] = (_Float16)a2[3];
        A[12] = (_Float16)a3[0]; A[13] = (_Float16)a3[1]; A[14] = (_Float16)a3[2]; A[15] = (_Float16)a3[3];

        const _Float16* bbase = sW + ((size_t)(kk << 2) * 32 + lane) * 16;
        const v16h B0 = *(const v16h*)(bbase + 0 * 512);
        const v16h B1 = *(const v16h*)(bbase + 1 * 512);
        const v16h B2 = *(const v16h*)(bbase + 2 * 512);
        const v16h B3 = *(const v16h*)(bbase + 3 * 512);

        acc0 = __builtin_amdgcn_wmma_f32_16x16x32_f16(false, A, false, B0, (short)0, acc0, false, false);
        acc1 = __builtin_amdgcn_wmma_f32_16x16x32_f16(false, A, false, B1, (short)0, acc1, false, false);
        acc2 = __builtin_amdgcn_wmma_f32_16x16x32_f16(false, A, false, B2, (short)0, acc2, false, false);
        acc3 = __builtin_amdgcn_wmma_f32_16x16x32_f16(false, A, false, B3, (short)0, acc3, false, false);
    }

    // ---- Fused epilogue: bias+mask, softmax, argmax, loss ----------------
    // D layout: lane<16 -> rows M=0..7 in acc[r], col=lane; lane>=16 -> M=8..15, col=lane-16.
    const int   texp = *texp_p;
    const float bc0 = bias[mrow], bc1 = bias[16 + mrow], bc2 = bias[32 + mrow], bc3 = bias[48 + mrow];
    const float kE1 = 2.7182818284590452f;  // exp(1)

    float lossAcc = 0.0f, rightAcc = 0.0f, cntAcc = 0.0f;

    #pragma unroll
    for (int r = 0; r < 8; ++r) {
        const int grow = row_base + hi * 8 + r;
        const float madd = (1.0f - amask[grow]) * -10000.0f;
        const float l0 = acc0[r] + bc0 + madd;
        const float l1 = acc1[r] + bc1 + madd;
        const float l2 = acc2[r] + bc2 + madd;
        const float l3 = acc3[r] + bc3 + madd;

        // local argmax (t ascending keeps first-occurrence semantics per lane)
        float mv = l0; int mi = mrow;
        if (l1 > mv) { mv = l1; mi = 16 + mrow; }
        if (l2 > mv) { mv = l2; mi = 32 + mrow; }
        if (l3 > mv) { mv = l3; mi = 48 + mrow; }
        #pragma unroll
        for (int off = 1; off <= 8; off <<= 1) {  // reduce within 16-lane half
            const float ov = __shfl_xor(mv, off, 32);
            const int   oi = __shfl_xor(mi, off, 32);
            if (ov > mv || (ov == mv && oi < mi)) { mv = ov; mi = oi; }
        }

        const float ex0 = expf(l0 - mv), ex1 = expf(l1 - mv);
        const float ex2 = expf(l2 - mv), ex3 = expf(l3 - mv);
        float s = ex0 + ex1 + ex2 + ex3;
        #pragma unroll
        for (int off = 1; off <= 8; off <<= 1) s += __shfl_xor(s, off, 32);
        const float inv = 1.0f / s;

        const int   tag = tags[grow];
        const float sc  = powf(t2s[tag], (float)texp);
        const float es  = expf(sc);
        const float Zy  = 63.0f * kE1 + es;
        const float yo  = kE1 / Zy, ytag = es / Zy;

        const float y0 = (mrow      == tag) ? ytag : yo;
        const float y1 = (16 + mrow == tag) ? ytag : yo;
        const float y2 = (32 + mrow == tag) ? ytag : yo;
        const float y3 = (48 + mrow == tag) ? ytag : yo;

        lossAcc -= y0 * logf(ex0 * inv + 1e-10f) + y1 * logf(ex1 * inv + 1e-10f)
                 + y2 * logf(ex2 * inv + 1e-10f) + y3 * logf(ex3 * inv + 1e-10f);

        if (mrow == 0) {  // one lane per row handles accuracy
            if (tag < kT - 3) {
                cntAcc += 1.0f;
                if (mi == tag) rightAcc += 1.0f;
            }
        }
    }

    // wave reduction + one atomic per wave
    #pragma unroll
    for (int off = 1; off < 32; off <<= 1) {
        lossAcc  += __shfl_xor(lossAcc,  off, 32);
        rightAcc += __shfl_xor(rightAcc, off, 32);
        cntAcc   += __shfl_xor(cntAcc,   off, 32);
    }
    if (lane == 0) {
        atomicAdd(&gacc[0], lossAcc);
        atomicAdd(&gacc[1], rightAcc);
        atomicAdd(&gacc[2], cntAcc);
    }
}

__global__ void mla_ner_fin(const float* __restrict__ g, float* __restrict__ out) {
    out[0] = g[0];
    out[1] = g[1] / g[2];
}

extern "C" void kernel_launch(void* const* d_in, const int* in_sizes, int n_in,
                              void* d_out, int out_size, void* d_ws, size_t ws_size,
                              hipStream_t stream) {
    const float* x     = (const float*)d_in[0];
    const float* W     = (const float*)d_in[1];
    const float* bias  = (const float*)d_in[2];
    const int*   tags  = (const int*)  d_in[3];
    const float* amask = (const float*)d_in[4];
    const float* t2s   = (const float*)d_in[5];
    const int*   texp  = (const int*)  d_in[6];
    float* gacc = (float*)d_ws;
    float* out  = (float*)d_out;

    const size_t lds_bytes = (size_t)kLdsHalves * sizeof(_Float16);  // 192 KB
    // Allow >64KB dynamic LDS (CDNA5 WGP supports up to 320KB/workgroup).
    (void)hipFuncSetAttribute((const void*)mla_ner_fused,
                              hipFuncAttributeMaxDynamicSharedMemorySize,
                              (int)lds_bytes);

    mla_ner_init<<<1, 32, 0, stream>>>(gacc);
    mla_ner_fused<<<kBlocks, 256, lds_bytes, stream>>>(x, W, bias, tags, amask, t2s, texp, gacc);
    mla_ner_fin<<<1, 1, 0, stream>>>(gacc, out);
}
